// AttentionBlock_2585570312360
// MI455X (gfx1250) — matrix operand order repaired and checked by reading, not compiled, not run on hardware
//
#include <hip/hip_runtime.h>
#include <hip/hip_bf16.h>

// Problem constants (fixed by reference): B=4, C=256, H=W=64 -> N=4096, 32 groups.
static constexpr int Bz = 4;
static constexpr int C  = 256;
static constexpr int N  = 4096;   // H*W
static constexpr int G  = 32;
static constexpr int CG = C / G;  // 8 channels per group

typedef __attribute__((ext_vector_type(16))) _Float16 v16h_t;
typedef __attribute__((ext_vector_type(8)))  _Float16 v8h_t;
typedef __attribute__((ext_vector_type(8)))  float    v8f_t;

union V16H { v16h_t v; v8h_t h[2]; };

// Build a 16-half WMMA operand fragment from two 8-half (128-bit) loads.
__device__ __forceinline__ v16h_t load_frag(const _Float16* p0, const _Float16* p1) {
    V16H u;
    u.h[0] = *(const v8h_t*)p0;
    u.h[1] = *(const v8h_t*)p1;
    return u.v;
}

__device__ __forceinline__ v8f_t wmma_f16(v16h_t a, v16h_t b, v8f_t c) {
    // (neg_a, A, neg_b, B, c_mod, C, reuse_a, reuse_b)
    return __builtin_amdgcn_wmma_f32_16x16x32_f16(false, a, false, b, (short)0, c,
                                                  false, false);
}

// ---------------------------------------------------------------------------
// Kernel 1: convert fp32 weights to f16 once.
// ---------------------------------------------------------------------------
__global__ void cvt_weights(const float* __restrict__ qw, const float* __restrict__ pw,
                            _Float16* __restrict__ wq, _Float16* __restrict__ wp) {
    int i = blockIdx.x * 256 + threadIdx.x;
    if (i < 3 * C * C) wq[i] = (_Float16)qw[i];
    if (i < C * C)     wp[i] = (_Float16)pw[i];
}

// ---------------------------------------------------------------------------
// Kernel 2: GroupNorm. One block per (batch, group). Writes h16 as [B][N][C]
// (token-major, channel-contiguous) so QKV B-fragments are vector loads.
// ---------------------------------------------------------------------------
__global__ void gn_kernel(const float* __restrict__ x, const float* __restrict__ gw,
                          const float* __restrict__ gb, _Float16* __restrict__ h16) {
    __shared__ float s1[256];
    __shared__ float s2[256];
    const int b = blockIdx.x >> 5;
    const int g = blockIdx.x & 31;
    const float* xp = x + (size_t)b * C * N + (size_t)(g * CG) * N;

    float sum = 0.f, ss = 0.f;
    for (int i = threadIdx.x; i < CG * N; i += 256) {
        float v = xp[i];
        sum += v; ss += v * v;
    }
    s1[threadIdx.x] = sum; s2[threadIdx.x] = ss;
    __syncthreads();
    for (int st = 128; st > 0; st >>= 1) {
        if (threadIdx.x < st) {
            s1[threadIdx.x] += s1[threadIdx.x + st];
            s2[threadIdx.x] += s2[threadIdx.x + st];
        }
        __syncthreads();
    }
    const float mean = s1[0] * (1.f / (CG * N));
    const float var  = s2[0] * (1.f / (CG * N)) - mean * mean;
    const float inv  = rsqrtf(var + 1e-5f);

    for (int i = threadIdx.x; i < CG * N; i += 256) {
        int cl = i >> 12;            // i / N
        int n  = i & (N - 1);
        int c  = g * CG + cl;
        float y = (xp[i] - mean) * inv * gw[c] + gb[c];
        h16[(size_t)b * N * C + (size_t)n * C + c] = (_Float16)y;
    }
}

// ---------------------------------------------------------------------------
// Kernel 3: QKV GEMM. qkv[o,n] = sum_c W[o,c] * h[c,n] + bias[o].
// Block = 4 waves; each wave does a 16(o) x 64(n) tile = 4 D-tiles.
// Routes outputs: q16,k16 as [B][N][C] ; v16 as [B][C][N].
// ---------------------------------------------------------------------------
__global__ void __launch_bounds__(128) qkv_gemm(
        const _Float16* __restrict__ w16, const float* __restrict__ bias,
        const _Float16* __restrict__ h16,
        _Float16* __restrict__ q16, _Float16* __restrict__ k16,
        _Float16* __restrict__ v16) {
    const int lane = threadIdx.x & 31;
    const int wave = threadIdx.x >> 5;
    const int hf   = lane >> 4;     // which half of the wave
    const int col  = lane & 15;
    const int b    = blockIdx.z;
    const int o0   = blockIdx.x * 64 + wave * 16;
    const int n0   = blockIdx.y * 64;

    v8f_t acc[4] = {};
    const _Float16* arow = w16 + (size_t)(o0 + col) * C;   // A row (M = col)
    const _Float16* hb   = h16 + (size_t)b * N * C;

    for (int kc = 0; kc < 8; ++kc) {
        const _Float16* ak = arow + kc * 32;
        v16h_t a = load_frag(ak + hf * 16, ak + hf * 16 + 8);
#pragma unroll
        for (int j = 0; j < 4; ++j) {
            const _Float16* bk = hb + (size_t)(n0 + j * 16 + col) * C + kc * 32 + hf * 16;
            v16h_t bf = load_frag(bk, bk + 8);
            acc[j] = wmma_f16(a, bf, acc[j]);
        }
    }
#pragma unroll
    for (int j = 0; j < 4; ++j) {
        int n_idx = n0 + j * 16 + col;
#pragma unroll
        for (int r = 0; r < 8; ++r) {
            int o_idx = o0 + r + hf * 8;
            float val = acc[j][r] + bias[o_idx];
            if (o_idx < C) {
                q16[(size_t)b * N * C + (size_t)n_idx * C + o_idx] = (_Float16)val;
            } else if (o_idx < 2 * C) {
                k16[(size_t)b * N * C + (size_t)n_idx * C + (o_idx - C)] = (_Float16)val;
            } else {
                v16[(size_t)b * C * N + (size_t)(o_idx - 2 * C) * N + n_idx] = (_Float16)val;
            }
        }
    }
}

// ---------------------------------------------------------------------------
// Kernel 4: flash attention. Block = 4 waves x 16 token-rows = 64 tokens.
// Per wave: Q (16x256) in registers, stream keys in chunks of 32, online
// softmax via half-wave shuffles, P restaged through a wave-private LDS slab,
// P·V accumulated into 16 D-tiles (16x256 f32). Output [B][N][C] f16.
// ---------------------------------------------------------------------------
__global__ void __launch_bounds__(128, 1) flash_kernel(
        const _Float16* __restrict__ q16, const _Float16* __restrict__ k16,
        const _Float16* __restrict__ v16, _Float16* __restrict__ o16) {
    __shared__ __align__(32) _Float16 plds[4][16][32];
    const int lane = threadIdx.x & 31;
    const int wave = threadIdx.x >> 5;
    const int hf   = lane >> 4;
    const int col  = lane & 15;
    const int b    = blockIdx.y;
    const int row0 = blockIdx.x * 64 + wave * 16;

    const _Float16* qb = q16 + (size_t)b * N * C;
    const _Float16* kb = k16 + (size_t)b * N * C;
    const _Float16* vb = v16 + (size_t)b * C * N;

    // Preload Q fragments (A operands): lane holds token row0+col.
    v16h_t qf[8];
    const _Float16* qrow = qb + (size_t)(row0 + col) * C;
#pragma unroll
    for (int kc = 0; kc < 8; ++kc) {
        const _Float16* p = qrow + kc * 32;
        qf[kc] = load_frag(p + hf * 16, p + hf * 16 + 8);
    }

    v8f_t oacc[16] = {};
    float rm[8], rl[8];
#pragma unroll
    for (int r = 0; r < 8; ++r) { rm[r] = -1e30f; rl[r] = 0.f; }
    const float scale = 0.0625f;  // C^-0.5 = 1/16

    for (int key0 = 0; key0 < N; key0 += 32) {
        if (key0 + 32 < N)
            __builtin_prefetch(kb + (size_t)(key0 + 32 + col) * C, 0, 1);

        // S = Q * K^T for 32 keys -> two 16x16 tiles.
        v8f_t s0 = {}, s1 = {};
#pragma unroll
        for (int kc = 0; kc < 8; ++kc) {
            const _Float16* k0p = kb + (size_t)(key0 + col) * C + kc * 32 + hf * 16;
            v16h_t b0 = load_frag(k0p, k0p + 8);
            s0 = wmma_f16(qf[kc], b0, s0);
            const _Float16* k1p = kb + (size_t)(key0 + 16 + col) * C + kc * 32 + hf * 16;
            v16h_t b1 = load_frag(k1p, k1p + 8);
            s1 = wmma_f16(qf[kc], b1, s1);
        }

        // Online softmax; D-layout row m = r + 8*hf lives in one 16-lane half.
#pragma unroll
        for (int r = 0; r < 8; ++r) {
            float a0 = s0[r] * scale;
            float a1 = s1[r] * scale;
            float vmax = fmaxf(a0, a1);
            vmax = fmaxf(vmax, __shfl_xor(vmax, 1));
            vmax = fmaxf(vmax, __shfl_xor(vmax, 2));
            vmax = fmaxf(vmax, __shfl_xor(vmax, 4));
            vmax = fmaxf(vmax, __shfl_xor(vmax, 8));
            float mnew = fmaxf(rm[r], vmax);
            float cf = __expf(rm[r] - mnew);
            float p0 = __expf(a0 - mnew);
            float p1 = __expf(a1 - mnew);
            float ps = p0 + p1;
            ps += __shfl_xor(ps, 1);
            ps += __shfl_xor(ps, 2);
            ps += __shfl_xor(ps, 4);
            ps += __shfl_xor(ps, 8);
            rl[r] = rl[r] * cf + ps;
            rm[r] = mnew;
            int m = r + hf * 8;
            plds[wave][m][col]      = (_Float16)p0;
            plds[wave][m][16 + col] = (_Float16)p1;
#pragma unroll
            for (int ct = 0; ct < 16; ++ct) oacc[ct][r] *= cf;
        }
        asm volatile("s_wait_dscnt 0" ::: "memory");

        // P as an A fragment (16x32), wave-private LDS slab, no cross-wave race.
        const _Float16* pr = &plds[wave][col][0];
        v16h_t pf = load_frag(pr + hf * 16, pr + hf * 16 + 8);

        // O += P * V : V chunk (32 keys x 256 ch) as 16 B fragments.
#pragma unroll
        for (int ct = 0; ct < 16; ++ct) {
            const _Float16* vp = vb + (size_t)(ct * 16 + col) * N + key0 + hf * 16;
            v16h_t bf = load_frag(vp, vp + 8);
            oacc[ct] = wmma_f16(pf, bf, oacc[ct]);
        }
    }

    // Normalize and write attention output as [B][N][C] (channel-contiguous).
#pragma unroll
    for (int r = 0; r < 8; ++r) {
        float inv = 1.f / rl[r];
        int token = row0 + r + hf * 8;
#pragma unroll
        for (int ct = 0; ct < 16; ++ct) {
            o16[(size_t)b * N * C + (size_t)token * C + ct * 16 + col] =
                (_Float16)(oacc[ct][r] * inv);
        }
    }
}

// ---------------------------------------------------------------------------
// Kernel 5: proj GEMM + bias + residual. out = x + Wp * attn + proj_b.
// ---------------------------------------------------------------------------
__global__ void __launch_bounds__(128) proj_gemm(
        const _Float16* __restrict__ w16, const float* __restrict__ bias,
        const _Float16* __restrict__ a16,   // attn out [B][N][C]
        const float* __restrict__ x, float* __restrict__ out) {
    const int lane = threadIdx.x & 31;
    const int wave = threadIdx.x >> 5;
    const int hf   = lane >> 4;
    const int col  = lane & 15;
    const int b    = blockIdx.z;
    const int o0   = blockIdx.x * 64 + wave * 16;
    const int n0   = blockIdx.y * 64;

    v8f_t acc[4] = {};
    const _Float16* arow = w16 + (size_t)(o0 + col) * C;
    const _Float16* ab   = a16 + (size_t)b * N * C;

    for (int kc = 0; kc < 8; ++kc) {
        const _Float16* ak = arow + kc * 32;
        v16h_t a = load_frag(ak + hf * 16, ak + hf * 16 + 8);
#pragma unroll
        for (int j = 0; j < 4; ++j) {
            const _Float16* bk = ab + (size_t)(n0 + j * 16 + col) * C + kc * 32 + hf * 16;
            v16h_t bf = load_frag(bk, bk + 8);
            acc[j] = wmma_f16(a, bf, acc[j]);
        }
    }
#pragma unroll
    for (int j = 0; j < 4; ++j) {
        int n_idx = n0 + j * 16 + col;
#pragma unroll
        for (int r = 0; r < 8; ++r) {
            int o_idx = o0 + r + hf * 8;
            size_t idx = (size_t)b * C * N + (size_t)o_idx * N + n_idx;
            out[idx] = x[idx] + acc[j][r] + bias[o_idx];
        }
    }
}

// ---------------------------------------------------------------------------
// Host entry
// ---------------------------------------------------------------------------
extern "C" void kernel_launch(void* const* d_in, const int* in_sizes, int n_in,
                              void* d_out, int out_size, void* d_ws, size_t ws_size,
                              hipStream_t stream) {
    const float* x      = (const float*)d_in[0];
    const float* gn_w   = (const float*)d_in[1];
    const float* gn_b   = (const float*)d_in[2];
    const float* qkv_w  = (const float*)d_in[3];
    const float* qkv_b  = (const float*)d_in[4];
    const float* proj_w = (const float*)d_in[5];
    const float* proj_b = (const float*)d_in[6];
    float* out = (float*)d_out;

    // Workspace carve-out (all chunks 256B-aligned by construction).
    char* w = (char*)d_ws;
    _Float16* wq  = (_Float16*)w; w += (size_t)3 * C * C * sizeof(_Float16); // 384 KB
    _Float16* wp  = (_Float16*)w; w += (size_t)C * C * sizeof(_Float16);     // 128 KB
    _Float16* buf = (_Float16*)w; w += (size_t)Bz * N * C * sizeof(_Float16); // h16, reused as attn-out
    _Float16* q16 = (_Float16*)w; w += (size_t)Bz * N * C * sizeof(_Float16);
    _Float16* k16 = (_Float16*)w; w += (size_t)Bz * N * C * sizeof(_Float16);
    _Float16* v16 = (_Float16*)w; w += (size_t)Bz * N * C * sizeof(_Float16);

    cvt_weights<<<(3 * C * C + 255) / 256, 256, 0, stream>>>(qkv_w, proj_w, wq, wp);
    gn_kernel<<<Bz * G, 256, 0, stream>>>(x, gn_w, gn_b, buf);
    qkv_gemm<<<dim3(3 * C / 64, N / 64, Bz), 128, 0, stream>>>(wq, qkv_b, buf,
                                                               q16, k16, v16);
    // flash writes into `buf` (h16 no longer needed after QKV GEMM)
    flash_kernel<<<dim3(N / 64, Bz), 128, 0, stream>>>(q16, k16, v16, buf);
    proj_gemm<<<dim3(C / 64, N / 64, Bz), 128, 0, stream>>>(wp, proj_b, buf, x, out);
}
